// RWKV6TimeMix_90228672955318
// MI455X (gfx1250) — matrix operand order repaired and checked by reading, not compiled, not run on hardware
//
#include <hip/hip_runtime.h>

// ---------------------------------------------------------------------------
// RWKV6 time-mix for MI455X (gfx1250, wave32, WMMA + async-to-LDS staging).
// B=4, T=1024, D=1024, H=16, DH=64
// ---------------------------------------------------------------------------

typedef __bf16 bf16;
typedef __attribute__((ext_vector_type(16))) __bf16 v16bf;
typedef __attribute__((ext_vector_type(8)))  float  v8f;

constexpr int  Bn  = 4;
constexpr int  Tn  = 1024;
constexpr int  Dn  = 1024;
constexpr int  Hn  = 16;
constexpr int  DHn = 64;
constexpr int  Mn  = Bn * Tn;                 // 4096 rows
constexpr size_t SX = (size_t)Mn * Dn;        // 4,194,304 elements

// ---------------------------------------------------------------------------
// CDNA5 async-to-LDS + raw split-barrier helpers (CDNA5 ISA, 01)
// ---------------------------------------------------------------------------
__device__ __forceinline__ void async_copy_b128(unsigned lds_addr, const bf16* gptr) {
  asm volatile("global_load_async_to_lds_b128 %0, %1, off"
               :: "v"(lds_addr), "v"((unsigned long long)(uintptr_t)gptr)
               : "memory");
}
__device__ __forceinline__ void wait_async0() { asm volatile("s_wait_asynccnt 0x0" ::: "memory"); }
__device__ __forceinline__ void wait_async1() { asm volatile("s_wait_asynccnt 0x1" ::: "memory"); }
// Execution-only workgroup barrier (no vmem drain): used right after
// s_wait_asynccnt to make the staged LDS tile visible to all waves.
__device__ __forceinline__ void wg_barrier() {
  asm volatile("s_barrier_signal -1\n\ts_barrier_wait -1" ::: "memory");
}
// Barrier that first drains this wave's outstanding LDS reads (DScnt only,
// loadcnt deliberately left in flight): protects the LDS buffer from being
// overwritten by the next phase's async copies while ds_loads are pending.
__device__ __forceinline__ void wg_barrier_ds() {
  asm volatile("s_wait_dscnt 0x0\n\ts_barrier_signal -1\n\ts_barrier_wait -1" ::: "memory");
}

// ---------------------------------------------------------------------------
// 1) Token-shift mix: xr/xk/xv/xw = tm*x + (1-tm)*x_prev, converted to bf16
// ---------------------------------------------------------------------------
__global__ __launch_bounds__(256)
void mix_kernel(const float* __restrict__ x,
                const float* __restrict__ tmr, const float* __restrict__ tmk,
                const float* __restrict__ tmv, const float* __restrict__ tmw,
                bf16* __restrict__ xr, bf16* __restrict__ xk,
                bf16* __restrict__ xv, bf16* __restrict__ xw) {
  size_t idx = (size_t)blockIdx.x * blockDim.x + threadIdx.x;
  if (idx >= SX) return;
  int d  = (int)(idx & (Dn - 1));
  size_t bt = idx >> 10;                      // idx / D
  int t  = (int)(bt & (Tn - 1));
  float xc = x[idx];
  float xp = (t == 0) ? 0.0f : x[idx - Dn];
  float ar = tmr[d], ak = tmk[d], av = tmv[d], aw = tmw[d];
  xr[idx] = (bf16)(ar * xc + (1.0f - ar) * xp);
  xk[idx] = (bf16)(ak * xc + (1.0f - ak) * xp);
  xv[idx] = (bf16)(av * xc + (1.0f - av) * xp);
  xw[idx] = (bf16)(aw * xc + (1.0f - aw) * xp);
}

// ---------------------------------------------------------------------------
// 2) Weight transpose + fp32->bf16: Wt[k][n] = W[n][k]  (D x D)
// ---------------------------------------------------------------------------
__global__ __launch_bounds__(1024)
void transpose_kernel(const float* __restrict__ W, bf16* __restrict__ Wt) {
  __shared__ float tile[32][33];
  int x = blockIdx.x * 32 + threadIdx.x;
  int y = blockIdx.y * 32 + threadIdx.y;
  tile[threadIdx.y][threadIdx.x] = W[(size_t)y * Dn + x];
  __syncthreads();
  int xo = blockIdx.y * 32 + threadIdx.x;     // n
  int yo = blockIdx.x * 32 + threadIdx.y;     // k
  Wt[(size_t)yo * Dn + xo] = (bf16)tile[threadIdx.x][threadIdx.y];
}

// ---------------------------------------------------------------------------
// 3) WMMA GEMM with async-LDS double buffering.
//    C[M x N] (fp32) = A[M x K] (bf16, row-major) @ Bt[K x N] (= A @ W^T).
//    Block: 256 thr (8 waves), tile 128M x 64N; wave -> 16M x 64N strip.
//    Per K-step(32): B tile 32x64 staged in LDS via
//    global_load_async_to_lds_b128 (double-buffered), A frags double-buffered
//    in registers; 4x v_wmma_f32_16x16x32_bf16 per wave per step.
// ---------------------------------------------------------------------------
union U16 { v16bf v; uint4 q[2]; };

__device__ __forceinline__ void load_a_frag(const bf16* pa, U16& a) {
  a.q[0] = *(const uint4*)(pa);
  a.q[1] = *(const uint4*)(pa + 16);
}

__device__ __forceinline__ void mma_step(const bf16* sbuf, int lane,
                                         const U16& a, v8f* acc) {
  // B frags (32x16 bf16 each): lane L holds column n=L%16, k in table order (8h+s | 16+8h+s-8).
  const int nl = lane & 15, hb = lane >> 4;
  U16 b0, b1, b2, b3;
#pragma unroll
  for (int s = 0; s < 16; ++s) {
    const int k = (s < 8) ? (8 * hb + s) : (16 + 8 * hb + (s - 8));
    const bf16* p = sbuf + k * 64 + nl;
    b0.v[s] = p[0]; b1.v[s] = p[16]; b2.v[s] = p[32]; b3.v[s] = p[48];
  }
  acc[0] = __builtin_amdgcn_wmma_f32_16x16x32_bf16(false, a.v, false, b0.v, (short)0, acc[0], false, false);
  acc[1] = __builtin_amdgcn_wmma_f32_16x16x32_bf16(false, a.v, false, b1.v, (short)0, acc[1], false, false);
  acc[2] = __builtin_amdgcn_wmma_f32_16x16x32_bf16(false, a.v, false, b2.v, (short)0, acc[2], false, false);
  acc[3] = __builtin_amdgcn_wmma_f32_16x16x32_bf16(false, a.v, false, b3.v, (short)0, acc[3], false, false);
}

__global__ __launch_bounds__(256)
void gemm_bf16_wmma(const bf16* __restrict__ A, const bf16* __restrict__ Bt,
                    float* __restrict__ C, int M, int N, int K) {
  __shared__ bf16 sB0[32 * 64];
  __shared__ bf16 sB1[32 * 64];

  const int tid  = threadIdx.x;
  const int lane = tid & 31;
  const int wave = tid >> 5;
  const int nblk = N >> 6;
  const int mB   = (blockIdx.x / nblk) * 128;   // block M origin
  const int n0   = (blockIdx.x % nblk) * 64;    // block N origin
  const int m0   = mB + wave * 16;              // wave strip

  // --- B staging: thread -> (row, 8 cols); one b128 async copy per K-step
  const int brow = tid >> 3;
  const int bcol = (tid & 7) * 8;
  const bf16* gB = Bt + (size_t)brow * N + n0 + bcol;
  const size_t gBstep = (size_t)32 * N;
  const unsigned lds0 = (unsigned)(uintptr_t)&sB0[brow * 64 + bcol];
  const unsigned lds1 = (unsigned)(uintptr_t)&sB1[brow * 64 + bcol];

  // --- A fragment pointers (16-bit A layout: two 8-elem runs per lane)
  const int mrow = m0 + (lane & 15);
  const int hs   = lane >> 4;
  const bf16* pA = A + (size_t)mrow * K + hs * 8;

  v8f acc[4];
#pragma unroll
  for (int s = 0; s < 4; ++s)
#pragma unroll
    for (int e = 0; e < 8; ++e) acc[s][e] = 0.0f;

  // Prologue: stage k-step 0
  async_copy_b128(lds0, gB);  gB += gBstep;
  U16 a0, a1;
  load_a_frag(pA, a0);        pA += 32;

  const int iters = K >> 6;                    // two 32-K-steps per iter
  for (int it = 0; it < iters; ++it) {
    // ---- phase 0: compute from sB0/a0, prefetch next step into sB1/a1
    async_copy_b128(lds1, gB);  gB += gBstep;
    load_a_frag(pA, a1);        pA += 32;      // stays in flight this phase
    wait_async1();              // sB0 resident (sB1 still in flight)
    wg_barrier();
    mma_step(sB0, lane, a0, acc);
    wg_barrier_ds();            // drain ds reads, then release sB0

    // ---- phase 1: compute from sB1/a1, prefetch next step into sB0/a0
    if (it + 1 < iters) {
      async_copy_b128(lds0, gB);  gB += gBstep;
      load_a_frag(pA, a0);        pA += 32;
      wait_async1();
    } else {
      wait_async0();
    }
    wg_barrier();
    mma_step(sB1, lane, a1, acc);
    wg_barrier_ds();            // drain ds reads, then release sB1
  }

  // C/D layout: VGPR r -> row m0 + r + 8*hs, col n0 + s*16 + (lane&15)
#pragma unroll
  for (int s = 0; s < 4; ++s) {
    int col = n0 + s * 16 + (lane & 15);
#pragma unroll
    for (int r = 0; r < 8; ++r) {
      int row = m0 + r + 8 * hs;
      C[(size_t)row * N + col] = acc[s][r];
    }
  }
}

// ---------------------------------------------------------------------------
// 4) Recurrent scan + fused GroupNorm + r-gating.
//    One workgroup per (b,h); 256 threads; thread (iseg,j) owns s[iseg*16..][j]
//    in 16 registers.  exp(w) = sigmoid(z)*exp(-1e-4), r = sigmoid(zr).
// ---------------------------------------------------------------------------
__global__ __launch_bounds__(256)
void scan_kernel(const float* __restrict__ rg, const float* __restrict__ kg,
                 const float* __restrict__ vg, const float* __restrict__ wg,
                 const float* __restrict__ u,  const float* __restrict__ lnw,
                 const float* __restrict__ lnb, bf16* __restrict__ on_out) {
  __shared__ float ew[DHn], ek[DHn], kk[DHn], rr[DHn], vv[DHn];
  __shared__ float uu[DHn], lw[DHn], lb[DHn], oo[DHn];
  __shared__ float part[4 * DHn];

  const int b = blockIdx.x / Hn;
  const int h = blockIdx.x % Hn;
  const int tid  = threadIdx.x;
  const int j    = tid & 63;
  const int iseg = tid >> 6;

  if (tid < DHn) {
    uu[tid] = u[h * DHn + tid];
    lw[tid] = lnw[tid];
    lb[tid] = lnb[tid];
  }
  __syncthreads();

  float s_reg[16];
#pragma unroll
  for (int ii = 0; ii < 16; ++ii) s_reg[ii] = 0.0f;

  for (int t = 0; t < Tn; ++t) {
    const size_t base = ((size_t)(b * Tn + t)) * Dn + h * DHn;
    if (tid < DHn) {
      int i = tid;
      float kvv = kg[base + i];
      float zw  = wg[base + i];
      float zr  = rg[base + i];
      ew[i] = 0.99990001f / (1.0f + __expf(-zw));   // exp(w_t)
      ek[i] = __expf(uu[i] + kvv);                  // exp(u + k_t)
      kk[i] = kvv;
      rr[i] = 1.0f / (1.0f + __expf(-zr));          // sigmoid(r)
      vv[i] = vg[base + i];
    }
    __syncthreads();

    const float vj = vv[j];
    float po = 0.0f;
#pragma unroll
    for (int ii = 0; ii < 16; ++ii) {
      const int i = iseg * 16 + ii;
      float bonus = ek[i] * vj;
      po += rr[i] * (s_reg[ii] + bonus);
      s_reg[ii] = ew[i] * s_reg[ii] + kk[i] * vj;
    }
    part[iseg * DHn + j] = po;
    __syncthreads();

    if (tid < DHn)
      oo[tid] = part[tid] + part[DHn + tid] + part[2 * DHn + tid] + part[3 * DHn + tid];
    __syncthreads();

    if (tid < DHn) {
      float mu = 0.0f;
#pragma unroll 8
      for (int q = 0; q < DHn; ++q) mu += oo[q];
      mu *= (1.0f / DHn);
      float var = 0.0f;
#pragma unroll 8
      for (int q = 0; q < DHn; ++q) { float dv = oo[q] - mu; var += dv * dv; }
      var *= (1.0f / DHn);
      float on = (oo[tid] - mu) * __frsqrt_rn(var + 1e-5f) * lw[tid] + lb[tid];
      on *= rr[tid];
      on_out[base + tid] = (bf16)on;
    }
    __syncthreads();
  }
}

// ---------------------------------------------------------------------------
// Orchestration
// ---------------------------------------------------------------------------
extern "C" void kernel_launch(void* const* d_in, const int* in_sizes, int n_in,
                              void* d_out, int out_size, void* d_ws, size_t ws_size,
                              hipStream_t stream) {
  const float* x    = (const float*)d_in[0];
  const float* W_r  = (const float*)d_in[1];
  const float* W_k  = (const float*)d_in[2];
  const float* W_v  = (const float*)d_in[3];
  const float* W_w  = (const float*)d_in[4];
  const float* W_o  = (const float*)d_in[5];
  const float* u    = (const float*)d_in[6];
  const float* tmr  = (const float*)d_in[7];
  const float* tmk  = (const float*)d_in[8];
  const float* tmv  = (const float*)d_in[9];
  const float* tmw  = (const float*)d_in[10];
  const float* lnw  = (const float*)d_in[11];
  const float* lnb  = (const float*)d_in[12];
  float* out = (float*)d_out;

  // Workspace layout
  char* ws = (char*)d_ws;
  size_t off = 0;
  bf16* xm[4];                                  // bf16 token-shift mixes
  for (int i = 0; i < 4; ++i) { xm[i] = (bf16*)(ws + off); off += SX * sizeof(bf16); }
  bf16* wt[5];                                  // bf16 transposed weights
  for (int i = 0; i < 5; ++i) { wt[i] = (bf16*)(ws + off); off += (size_t)Dn * Dn * sizeof(bf16); }
  float* proj[4];                               // fp32 GEMM outputs r,k,v,w
  for (int i = 0; i < 4; ++i) { proj[i] = (float*)(ws + off); off += SX * sizeof(float); }
  bf16* on_bf = (bf16*)(ws + off); off += SX * sizeof(bf16);

  // 1) token-shift mixes -> bf16
  mix_kernel<<<(int)(SX / 256), 256, 0, stream>>>(x, tmr, tmk, tmv, tmw,
                                                  xm[0], xm[1], xm[2], xm[3]);

  // 2) weight transposes -> bf16
  {
    dim3 blk(32, 32), grd(Dn / 32, Dn / 32);
    const float* Ws[5] = {W_r, W_k, W_v, W_w, W_o};
    for (int i = 0; i < 5; ++i)
      transpose_kernel<<<grd, blk, 0, stream>>>(Ws[i], wt[i]);
  }

  // 3) four projection GEMMs (M=4096, N=K=1024) via bf16 WMMA
  {
    int blocks = (Mn / 128) * (Dn / 64);        // 512 blocks of 8 waves
    for (int i = 0; i < 4; ++i)
      gemm_bf16_wmma<<<blocks, 256, 0, stream>>>(xm[i], wt[i], proj[i], Mn, Dn, Dn);
  }

  // 4) recurrent scan + groupnorm + gating -> bf16 o_n
  scan_kernel<<<Bn * Hn, 256, 0, stream>>>(proj[0], proj[1], proj[2], proj[3],
                                           u, lnw, lnb, on_bf);

  // 5) output projection: out = o_n @ W_o^T   (fp32 result)
  {
    int blocks = (Mn / 128) * (Dn / 64);
    gemm_bf16_wmma<<<blocks, 256, 0, stream>>>(on_bf, wt[4], out, Mn, Dn, Dn);
  }
  (void)in_sizes; (void)n_in; (void)out_size; (void)ws_size;
}
